// Gauss_Process_55843164783486
// MI455X (gfx1250) — matrix operand order repaired and checked by reading, not compiled, not run on hardware
//
#include <hip/hip_runtime.h>
#include <hip/hip_bf16.h>

// ---------------------------------------------------------------------------
// SNGP / GP head on MI455X (gfx1250): three bf16 WMMA GEMMs with fused
// epilogues.  Compute-bound (~120 GFLOP vs ~90MB traffic) -> drive the
// v_wmma_f32_16x16x32_bf16 pipe.  Block = 8 waves sharing a 64-column group;
// the B panel is staged into LDS once per block per 64-deep k-chunk with the
// CDNA5 async DMA path (global_load_async_to_lds_b128 / s_wait_asynccnt),
// double-buffered.  A fragments are per-wave register-double-buffered global
// b128 loads.  Per-wave register tile: 32x64 (8 f32 accumulators).
// ---------------------------------------------------------------------------

typedef __attribute__((ext_vector_type(16))) __bf16 v16bf;
typedef __attribute__((ext_vector_type(8)))  float  v8f;
typedef unsigned long long ull;

#define IN_F    512
#define RFF     2048
#define OUT_F   1000
#define OUT_PAD 1024
#define NROWS   8192
#define INV_RFF_SCALAR 0.03125f   // 1/sqrt(2048/2)
#define MFF     25.0f

// LDS B panel: 64 columns x 64 halves per chunk, padded to 72 halves/row so
// the 16 consuming lanes map to 16 distinct bank groups.  Two buffers.
#define BROW    72
#define BBUF_H  (64 * BROW)          // halves per buffer
#define BBUF_B  (BBUF_H * 2)         // bytes per buffer (9216)

union Frag {
    v16bf v;
    unsigned short s[16];
    ull d[4];
};

__device__ __forceinline__ unsigned short f2bf(float f) {
    unsigned u = __float_as_uint(f);
    u += 0x7FFFu + ((u >> 16) & 1u);          // round-to-nearest-even
    return (unsigned short)(u >> 16);
}
__device__ __forceinline__ float bf2f(unsigned short h) {
    return __uint_as_float(((unsigned)h) << 16);
}

// ---------------- CDNA5 async global->LDS DMA (ASYNCcnt) -------------------
// The VDST operand of global_load_async_to_lds_b128 is the wave-relative LDS
// byte offset; the low 32 bits of a flat pointer to __shared__ memory are
// exactly that offset (ISA 10.2).  Deriving the
// offset from the array's address (ptrtoint) also makes the array escape, so
// the "memory" clobber keeps the later ds_loads of the staged panel alive.
__device__ __forceinline__ void async_b128(unsigned short* lds_dst,
                                           const unsigned short* g) {
    unsigned loff = (unsigned)(uintptr_t)lds_dst;
    asm volatile("global_load_async_to_lds_b128 %0, %1, off"
                 :: "v"(loff), "v"((ull)(uintptr_t)g)
                 : "memory");
}
__device__ __forceinline__ void wait_async0() {
    asm volatile("s_wait_asynccnt 0x0" ::: "memory");
}

// -------------------------- conversion kernels -----------------------------

__global__ void __launch_bounds__(256)
k_cvt_copy(const float* __restrict__ src, unsigned short* __restrict__ dst, int n) {
    int i = blockIdx.x * blockDim.x + threadIdx.x;
    if (i < n) dst[i] = f2bf(src[i]);
}

// dst[n*K + k] = src[k*N + n]   (src is K x N row-major)
__global__ void __launch_bounds__(256)
k_cvt_transpose(const float* __restrict__ src, unsigned short* __restrict__ dst,
                int K, int N) {
    int i = blockIdx.x * blockDim.x + threadIdx.x;
    if (i < K * N) {
        int n = i / K;
        int k = i % K;
        dst[i] = f2bf(src[k * N + n]);
    }
}

// copy rows [0, rows_src) and zero-pad rows [rows_src, rows_dst)
__global__ void __launch_bounds__(256)
k_cvt_pad_rows(const float* __restrict__ src, unsigned short* __restrict__ dst,
               int rows_src, int rows_dst, int cols) {
    int i = blockIdx.x * blockDim.x + threadIdx.x;
    if (i < rows_dst * cols) {
        int r = i / cols;
        dst[i] = (r < rows_src) ? f2bf(src[i]) : (unsigned short)0;
    }
}

__global__ void __launch_bounds__(256)
k_zero_f32(float* __restrict__ p, int n) {
    int i = blockIdx.x * blockDim.x + threadIdx.x;
    if (i < n) p[i] = 0.0f;
}

// -------------------------- WMMA fragment loads ----------------------------
// A row-major [M, lda] bf16.  16x32 bf16 A layout per lane-half h:
//   element j=0..7  -> K = k + 8h + j        (contiguous 16B)
//   element j=8..15 -> K = k + 16 + 8h + (j-8)  (contiguous 16B)
__device__ __forceinline__ void load_a(Frag& a, const unsigned short* arow,
                                       int k, int h) {
    const ull* p0 = (const ull*)(arow + k + h * 8);
    const ull* p1 = (const ull*)(arow + k + 16 + h * 8);
    a.d[0] = p0[0]; a.d[1] = p0[1];
    a.d[2] = p1[0]; a.d[3] = p1[1];
}

// Stage one 64-deep B chunk (64 cols x 128B) into an LDS buffer with async
// DMA: 512 x 16B transfers, 2 per thread of the 256-thread block.
__device__ __forceinline__ void stage_b(const unsigned short* __restrict__ Bt,
                                        int ldb, int n0, int kc,
                                        unsigned short* __restrict__ buf,
                                        int tid) {
#pragma unroll
    for (int e2 = 0; e2 < 2; ++e2) {
        const int e   = tid + e2 * 256;
        const int col = e >> 3;
        const int seg = e & 7;
        const unsigned short* g = Bt + (size_t)(n0 + col) * ldb + kc + seg * 8;
        async_b128(buf + col * BROW + seg * 8, g);
    }
}

// 16 WMMAs of one 64-deep chunk: B fragments read from the LDS buffer.
// LDS B layout: column c at c*BROW halves, K-halves contiguous; fragment for
// lane-half h, k-substep ks: 8 halves at ks*32 + h*8 and 8 at ks*32 + 16 + h*8.
__device__ __forceinline__ void compute_chunk(
    const unsigned short* __restrict__ Bbuf, const Frag aF[2][2],
    v8f acc[2][4], int h, int ln) {
#pragma unroll
    for (int ks = 0; ks < 2; ++ks) {
#pragma unroll
        for (int t = 0; t < 4; ++t) {
            Frag b;
            const ull* q = (const ull*)
                (Bbuf + (size_t)(t * 16 + ln) * BROW + ks * 32 + h * 8);
            b.d[0] = q[0]; b.d[1] = q[1]; b.d[2] = q[4]; b.d[3] = q[5];
#pragma unroll
            for (int i = 0; i < 2; ++i)
                acc[i][t] = __builtin_amdgcn_wmma_f32_16x16x32_bf16(
                    false, aF[i][ks].v, false, b.v, (short)0, acc[i][t],
                    false, false);
        }
    }
}

// ---------------- block-cooperative 256x64 k-loop --------------------------
// Requires K % 128 == 0 (true for 512 and 2048).
__device__ __forceinline__ void wmma_loop_coop(
    const unsigned short* __restrict__ A, int lda,
    const unsigned short* __restrict__ Bt, int ldb,
    int K, int m0, int n0, int tid, unsigned short* Blds, v8f acc[2][4]) {
    const int lane = tid & 31;
    const int h    = lane >> 4;
    const int ln   = lane & 15;
    const unsigned short* arow[2];
    arow[0] = A + (size_t)(m0 + ln) * lda;
    arow[1] = A + (size_t)(m0 + 16 + ln) * lda;

    Frag aS0[2][2], aS1[2][2];
    // prime: A for chunk 0, async-stage chunk 0 into buffer 0
#pragma unroll
    for (int i = 0; i < 2; ++i)
#pragma unroll
        for (int ks = 0; ks < 2; ++ks)
            load_a(aS0[i][ks], arow[i], ks * 32, h);
    stage_b(Bt, ldb, n0, 0, Blds, tid);
    wait_async0();
    __syncthreads();

    for (int kc = 0; kc < K; kc += 128) {
        // stage chunk kc+64 into buffer 1, prefetch its A fragments
        stage_b(Bt, ldb, n0, kc + 64, Blds + BBUF_H, tid);
#pragma unroll
        for (int i = 0; i < 2; ++i)
#pragma unroll
            for (int ks = 0; ks < 2; ++ks)
                load_a(aS1[i][ks], arow[i], kc + 64 + ks * 32, h);
        // compute chunk kc from buffer 0
        compute_chunk(Blds, aS0, acc, h, ln);
        wait_async0();
        __syncthreads();

        if (kc + 128 < K) {   // stage chunk kc+128 back into buffer 0
            stage_b(Bt, ldb, n0, kc + 128, Blds, tid);
#pragma unroll
            for (int i = 0; i < 2; ++i)
#pragma unroll
                for (int ks = 0; ks < 2; ++ks)
                    load_a(aS0[i][ks], arow[i], kc + 128 + ks * 32, h);
        }
        // compute chunk kc+64 from buffer 1
        compute_chunk(Blds + BBUF_H, aS1, acc, h, ln);
        wait_async0();
        __syncthreads();
    }
}

// ---------------- GEMM1: Phi = cos(D @ W + b) / 32  (bf16 out) -------------
__global__ void __launch_bounds__(256)
k_gemm_rff(const unsigned short* __restrict__ Dh,
           const unsigned short* __restrict__ Wt,
           const float* __restrict__ bvec,
           unsigned short* __restrict__ Phi) {
    __shared__ unsigned short Blds[2 * BBUF_H];
    const int tid  = threadIdx.x;
    const int NT4  = RFF / 64;                         // 32 column groups
    const int m0   = (blockIdx.x / NT4) * 256 + (tid >> 5) * 32;
    const int n0   = (blockIdx.x % NT4) * 64;
    v8f acc[2][4] = {};
    wmma_loop_coop(Dh, IN_F, Wt, IN_F, IN_F, m0, n0, tid, Blds, acc);
    const int h = (tid >> 4) & 1, ln = tid & 15;
#pragma unroll
    for (int t = 0; t < 4; ++t) {
        const int n = n0 + t * 16 + ln;
        const float bb = bvec[n];
#pragma unroll
        for (int i = 0; i < 2; ++i)
#pragma unroll
            for (int r = 0; r < 8; ++r) {
                const int m = m0 + i * 16 + r + 8 * h;
                const float z = acc[i][t][r] + bb;
                Phi[(size_t)m * RFF + n] = f2bf(__cosf(z) * INV_RFF_SCALAR);
            }
    }
}

// -------- GEMM3 fused: diag[m] += sum_n (Phi@cov)[m,n] * Phi[m,n] ----------
__global__ void __launch_bounds__(256)
k_gemm_diag(const unsigned short* __restrict__ Phi,
            const unsigned short* __restrict__ covT,
            float* __restrict__ diag) {
    __shared__ unsigned short Blds[2 * BBUF_H];
    const int tid  = threadIdx.x;
    const int NT4  = RFF / 64;
    const int m0   = (blockIdx.x / NT4) * 256 + (tid >> 5) * 32;
    const int n0   = (blockIdx.x % NT4) * 64;
    v8f acc[2][4] = {};
    wmma_loop_coop(Phi, RFF, covT, RFF, RFF, m0, n0, tid, Blds, acc);
    const int h = (tid >> 4) & 1, ln = tid & 15;
    float part[2][8] = {};
#pragma unroll
    for (int t = 0; t < 4; ++t) {
        const int n = n0 + t * 16 + ln;
#pragma unroll
        for (int i = 0; i < 2; ++i)
#pragma unroll
            for (int r = 0; r < 8; ++r) {
                const int m = m0 + i * 16 + r + 8 * h;
                part[i][r] += acc[i][t][r] * bf2f(Phi[(size_t)m * RFF + n]);
            }
    }
    // reduce over the 16 lanes sharing a column group (xor stays in-half)
#pragma unroll
    for (int mask = 1; mask <= 8; mask <<= 1)
#pragma unroll
        for (int i = 0; i < 2; ++i)
#pragma unroll
            for (int r = 0; r < 8; ++r)
                part[i][r] += __shfl_xor(part[i][r], mask, 32);
    if (ln == 0) {
#pragma unroll
        for (int i = 0; i < 2; ++i)
#pragma unroll
            for (int r = 0; r < 8; ++r)
                atomicAdd(&diag[m0 + i * 16 + r + 8 * h], part[i][r]);
    }
}

// ---- GEMM2 fused: out = (Phi @ lw^T + lb) * rsqrt(1 + 25*diag) ------------
__global__ void __launch_bounds__(256)
k_gemm_out(const unsigned short* __restrict__ Phi,
           const unsigned short* __restrict__ LWh,
           const float* __restrict__ logit_b,
           const float* __restrict__ diag,
           float* __restrict__ out) {
    __shared__ unsigned short Blds[2 * BBUF_H];
    const int tid  = threadIdx.x;
    const int NT4  = OUT_PAD / 64;                     // 16 column groups
    const int m0   = (blockIdx.x / NT4) * 256 + (tid >> 5) * 32;
    const int n0   = (blockIdx.x % NT4) * 64;
    v8f acc[2][4] = {};
    wmma_loop_coop(Phi, RFF, LWh, RFF, RFF, m0, n0, tid, Blds, acc);
    const int h = (tid >> 4) & 1, ln = tid & 15;
    float sc[2][8];
#pragma unroll
    for (int i = 0; i < 2; ++i)
#pragma unroll
        for (int r = 0; r < 8; ++r)
            sc[i][r] = rsqrtf(1.0f + MFF * diag[m0 + i * 16 + r + 8 * h]);
#pragma unroll
    for (int t = 0; t < 4; ++t) {
        const int n = n0 + t * 16 + ln;
        if (n < OUT_F) {
            const float lb = logit_b[n];
#pragma unroll
            for (int i = 0; i < 2; ++i)
#pragma unroll
                for (int r = 0; r < 8; ++r) {
                    const int m = m0 + i * 16 + r + 8 * h;
                    out[(size_t)m * OUT_F + n] = (acc[i][t][r] + lb) * sc[i][r];
                }
        }
    }
}

// --------------------------------- launch ----------------------------------

extern "C" void kernel_launch(void* const* d_in, const int* in_sizes, int n_in,
                              void* d_out, int out_size, void* d_ws, size_t ws_size,
                              hipStream_t stream) {
    const float* D   = (const float*)d_in[0];   // [8192, 512]
    const float* W   = (const float*)d_in[1];   // [512, 2048]
    const float* b   = (const float*)d_in[2];   // [2048]
    const float* lw  = (const float*)d_in[3];   // [1000, 2048]
    const float* lb  = (const float*)d_in[4];   // [1000]
    const float* cov = (const float*)d_in[5];   // [2048, 2048]
    float* out = (float*)d_out;                 // [8192, 1000]

    char* ws = (char*)d_ws;
    unsigned short* Dh   = (unsigned short*)(ws);                       //  8.00 MB
    unsigned short* Wt   = (unsigned short*)(ws +  8388608);            //  2.00 MB
    unsigned short* LWh  = (unsigned short*)(ws + 10485760);            //  4.00 MB
    unsigned short* covT = (unsigned short*)(ws + 14680064);            //  8.00 MB
    unsigned short* Phi  = (unsigned short*)(ws + 23068672);            // 32.00 MB
    float*          diag = (float*)(ws + 56623104);                     // 32 KB

    const dim3 blk(256);
    // precision conversion / layout staging (bf16, B matrices transposed)
    k_cvt_copy     <<<(NROWS * IN_F + 255) / 256, blk, 0, stream>>>(D, Dh, NROWS * IN_F);
    k_cvt_transpose<<<(RFF * IN_F + 255) / 256,  blk, 0, stream>>>(W, Wt, IN_F, RFF);
    k_cvt_pad_rows <<<(OUT_PAD * RFF + 255) / 256, blk, 0, stream>>>(lw, LWh, OUT_F, OUT_PAD, RFF);
    k_cvt_transpose<<<(RFF * RFF + 255) / 256,   blk, 0, stream>>>(cov, covT, RFF, RFF);
    k_zero_f32     <<<(NROWS + 255) / 256,       blk, 0, stream>>>(diag, NROWS);

    // GEMM1: (8192/256) row blocks x (2048/64) col groups = 1024 blocks
    k_gemm_rff <<<1024, 256, 0, stream>>>(Dh, Wt, b, Phi);
    // GEMM3 fused diag: same tiling, K = 2048
    k_gemm_diag<<<1024, 256, 0, stream>>>(Phi, covT, diag);
    // GEMM2 fused epilogue: 32 x 16 = 512 blocks
    k_gemm_out <<<512, 256, 0, stream>>>(Phi, LWh, lb, diag, out);
}
